// FullAttention_ablation_9749575762797
// MI455X (gfx1250) — matrix operand order repaired and checked by reading, not compiled, not run on hardware
//
#include <hip/hip_runtime.h>
#include <hip/hip_bf16.h>
#include <math.h>

// ---------------------------------------------------------------------------
// FullAttention ablation: O = softmax((QK^T/sqrt(E)) * wm) V
//   wm = scale * softmax(weight_mat / softplus(tau))  -> packed C-frag layout
//   Q/K/V pre-packed to bf16 in WMMA-fragment layout  (once, in ws)
// Hot loop: v_wmma_f32_16x16x32_bf16, M=32 query rows per wave sharing K/V
// fragments; all hot-loop loads are aligned 32B vector loads (b128 pairs).
// B=8, L=S=T=1024, H=8, E=64.
// ---------------------------------------------------------------------------

typedef __attribute__((ext_vector_type(16))) __bf16 v16bf;
typedef __attribute__((ext_vector_type(2)))  __bf16 v2bf;
typedef __attribute__((ext_vector_type(8)))  float  v8f;

#define B_DIM 8
#define L_DIM 1024
#define H_DIM 8
#define E_DIM 64
#define WAVES_PER_BLOCK 4

// workspace byte offsets
#define WM_OFF  (0)                           // 64*64 tiles * 256 f32 = 4 MB
#define Q_OFF   (4u  * 1024 * 1024)           // 64*64*2 frags bf16    = 8 MB
#define K_OFF   (12u * 1024 * 1024)           // 64*64*2 frags bf16    = 8 MB
#define V_OFF   (20u * 1024 * 1024)           // 64*32*4 frags bf16    = 8 MB

__device__ __forceinline__ size_t gidx(int b, int l, int h, int e) {
  return (((size_t)b * L_DIM + l) * H_DIM + h) * E_DIM + e;
}

// -------------------------------------------------------------------
// Kernel 1: wmp = (1/sqrt(E)) * softmax(weight_mat / softplus(tau)),
// written directly in C-fragment tile layout:
//   wmp[((rowTile*64 + colTile)*32 + lane)*8 + r]
//     = scale * wm[rowTile*16 + (lane>>4)*8 + r][colTile*16 + (lane&15)]
// so the attention kernel reads one aligned v8f per lane per 16x16 tile.
// -------------------------------------------------------------------
__global__ __launch_bounds__(256)
void wm_softmax_kernel(const float* __restrict__ w,
                       const float* __restrict__ tau,
                       float* __restrict__ wmp) {
  const int row = blockIdx.x;
  const int tid = threadIdx.x;
  const float t   = tau[row];                 // tau is [1,1,T,1] -> flat T
  const float inv = 1.0f / log1pf(__expf(t)); // 1/softplus(tau)
  const float* wr = w + (size_t)row * L_DIM;

  __shared__ float red[256];

  float mx = -INFINITY;
  for (int i = tid; i < L_DIM; i += 256) mx = fmaxf(mx, wr[i] * inv);
  red[tid] = mx; __syncthreads();
  for (int s = 128; s > 0; s >>= 1) {
    if (tid < s) red[tid] = fmaxf(red[tid], red[tid + s]);
    __syncthreads();
  }
  mx = red[0]; __syncthreads();

  float sum = 0.0f;
  for (int i = tid; i < L_DIM; i += 256) sum += __expf(wr[i] * inv - mx);
  red[tid] = sum; __syncthreads();
  for (int s = 128; s > 0; s >>= 1) {
    if (tid < s) red[tid] += red[tid + s];
    __syncthreads();
  }
  const float rs = 0.125f / red[0];           // fold in 1/sqrt(64)

  const int rt = row >> 4, rr = row & 15;
  const int khh = rr >> 3, r = rr & 7;
  for (int i = tid; i < L_DIM; i += 256) {
    const int st = i >> 4, n = i & 15;
    const size_t o = (((size_t)(rt * 64 + st) * 32) + (khh * 16 + n)) * 8 + r;
    wmp[o] = __expf(wr[i] * inv - mx) * rs;
  }
}

// -------------------------------------------------------------------
// Pack kernels: f32 [B,L,H,E] -> bf16 WMMA fragments.
// One thread per packed bf16 pair (one 32-bit store).
// Fragment = 32 lanes x 16 bf16 (512 elems); pair index j in [0,8).
// -------------------------------------------------------------------

// Q as A-fragments (16x32): f = (bh*64 + rowTile)*2 + eHalf
__global__ __launch_bounds__(256)
void packQ_kernel(const float* __restrict__ Q, v2bf* __restrict__ qbf) {
  const unsigned gid  = blockIdx.x * 256 + threadIdx.x;
  const int j    = gid & 7;
  const int lane = (gid >> 3) & 31;
  const int f    = gid >> 8;
  const int eh   = f & 1;
  const int rt   = (f >> 1) & 63;
  const int bh   = f >> 7;
  const int b = bh >> 3, h = bh & 7;
  const int m = lane & 15, kh = lane >> 4;
  const int kp = kh * 16 + 2 * j;
  const int e  = eh * 32 + kp;
  const int l  = rt * 16 + m;
  v2bf p;
  p[0] = (__bf16)Q[gidx(b, l, h, e)];
  p[1] = (__bf16)Q[gidx(b, l, h, e + 1)];
  qbf[gid] = p;
}

// K as B-fragments (32x16, K-dim = e): f = (bh*64 + sTile)*2 + eHalf
__global__ __launch_bounds__(256)
void packK_kernel(const float* __restrict__ K, v2bf* __restrict__ kbf) {
  const unsigned gid  = blockIdx.x * 256 + threadIdx.x;
  const int j    = gid & 7;
  const int lane = (gid >> 3) & 31;
  const int f    = gid >> 8;
  const int eh   = f & 1;
  const int st   = (f >> 1) & 63;
  const int bh   = f >> 7;
  const int b = bh >> 3, h = bh & 7;
  const int n = lane & 15, kh = lane >> 4;
  const int e = eh * 32 + kh * 16 + 2 * j;
  const int s = st * 16 + n;
  v2bf p;
  p[0] = (__bf16)K[gidx(b, s, h, e)];
  p[1] = (__bf16)K[gidx(b, s, h, e + 1)];
  kbf[gid] = p;
}

// V as B-fragments (32x16, K-dim = s): f = (bh*32 + sStep)*4 + colTile
__global__ __launch_bounds__(256)
void packV_kernel(const float* __restrict__ V, v2bf* __restrict__ vbf) {
  const unsigned gid  = blockIdx.x * 256 + threadIdx.x;
  const int j    = gid & 7;
  const int lane = (gid >> 3) & 31;
  const int f    = gid >> 8;
  const int ct   = f & 3;
  const int ss   = (f >> 2) & 31;
  const int bh   = f >> 7;
  const int b = bh >> 3, h = bh & 7;
  const int n = lane & 15, kh = lane >> 4;
  const int s = ss * 32 + kh * 16 + 2 * j;
  const int d = ct * 16 + n;
  v2bf p;
  p[0] = (__bf16)V[gidx(b, s,     h, d)];
  p[1] = (__bf16)V[gidx(b, s + 1, h, d)];
  vbf[gid] = p;
}

// -------------------------------------------------------------------
// Kernel 3: fused flash-style attention.
// One wave32 owns a 32-row query tile (two 16x16 M-tiles) of one (b,h)
// and shares each loaded K/V fragment across both M-tiles. 2048 waves.
// -------------------------------------------------------------------
__global__ __launch_bounds__(32 * WAVES_PER_BLOCK)
void attn_kernel(const v16bf* __restrict__ qbf,
                 const v16bf* __restrict__ kbf,
                 const v16bf* __restrict__ vbf,
                 const float* __restrict__ wmp,
                 float* __restrict__ out) {
  const int lane      = threadIdx.x & 31;
  const int waveInBlk = threadIdx.x >> 5;
  const int wave      = blockIdx.x * WAVES_PER_BLOCK + waveInBlk;

  const int wt   = wave & 31;           // 32 row-pair tiles per (b,h)
  const int bh   = wave >> 5;
  const int b    = bh >> 3;             // H = 8
  const int h    = bh & 7;
  const int row0 = wt * 32;

  const int m  = lane & 15;             // N column (B/C/D) / M row (A)
  const int kh = lane >> 4;             // lane half

  // Q A-fragments for both 16-row M-tiles, both e-halves (4 x 32B loads)
  v16bf aq[2][2];
#pragma unroll
  for (int mt = 0; mt < 2; ++mt)
#pragma unroll
    for (int eh = 0; eh < 2; ++eh)
      aq[mt][eh] = qbf[((bh * 64 + wt * 2 + mt) * 2 + eh) * 32 + lane];

  // all-ones B fragment: P row sums on the tensor pipe
  v16bf bones;
#pragma unroll
  for (int i = 0; i < 16; ++i) bones[i] = (__bf16)1.0f;

  // acc[mt][0..3] = output column tiles, acc[mt][4] = softmax denominator
  v8f acc[2][5] = {{v8f{}, v8f{}, v8f{}, v8f{}, v8f{}},
                   {v8f{}, v8f{}, v8f{}, v8f{}, v8f{}}};
  float mrow[2][8];
#pragma unroll
  for (int mt = 0; mt < 2; ++mt)
#pragma unroll
    for (int r = 0; r < 8; ++r) mrow[mt][r] = -INFINITY;

  // per-wave LDS tiles for P C-layout -> A-layout transpose (2 x 16x32 bf16)
  __shared__ unsigned int ldsP32[WAVES_PER_BLOCK][2][16 * 16];

  const v16bf* kbase = kbf + (size_t)(bh * 64) * 2 * 32;
  const v16bf* vbase = vbf + (size_t)(bh * 32) * 4 * 32;
  // per-M-tile wm bases; tile row rt = wt*2 + mt, each tile = 256 f32 (1 KB)
  const float* wbase0 = wmp + (((size_t)(wt * 2 + 0) * 64) * 32 + lane) * 8;
  const float* wbase1 = wmp + (((size_t)(wt * 2 + 1) * 64) * 32 + lane) * 8;

  for (int sbase = 0; sbase < L_DIM; sbase += 32) {
    const int st = sbase >> 4;          // first 16-col tile index
    const int ss = sbase >> 5;          // 32-col step index

    // prefetch next step's fragments
    if (sbase + 32 < L_DIM) {
      __builtin_prefetch(&kbase[((st + 2) * 2) * 32 + lane], 0, 1);
      __builtin_prefetch(&vbase[((ss + 1) * 4) * 32 + lane], 0, 1);
    }

    // ---- K fragments, shared by both M-tiles ------------------------
    const v16bf bk00 = kbase[((st    ) * 2 + 0) * 32 + lane];
    const v16bf bk01 = kbase[((st    ) * 2 + 1) * 32 + lane];
    const v16bf bk10 = kbase[((st + 1) * 2 + 0) * 32 + lane];
    const v16bf bk11 = kbase[((st + 1) * 2 + 1) * 32 + lane];

    // ---- wm tiles (scale pre-folded), one aligned v8f per tile ------
    const v8f wv00 = *(const v8f*)(wbase0 + (size_t)(st    ) * 256);
    const v8f wv01 = *(const v8f*)(wbase0 + (size_t)(st + 1) * 256);
    const v8f wv10 = *(const v8f*)(wbase1 + (size_t)(st    ) * 256);
    const v8f wv11 = *(const v8f*)(wbase1 + (size_t)(st + 1) * 256);

    // ---- per M-tile: scores, *wm, online softmax, stage P -----------
#pragma unroll
    for (int mt = 0; mt < 2; ++mt) {
      v8f c0 = {}, c1 = {};
      c0 = __builtin_amdgcn_wmma_f32_16x16x32_bf16(false, aq[mt][0], false, bk00,
                                                   (short)0, c0, false, false);
      c0 = __builtin_amdgcn_wmma_f32_16x16x32_bf16(false, aq[mt][1], false, bk01,
                                                   (short)0, c0, false, false);
      c1 = __builtin_amdgcn_wmma_f32_16x16x32_bf16(false, aq[mt][0], false, bk10,
                                                   (short)0, c1, false, false);
      c1 = __builtin_amdgcn_wmma_f32_16x16x32_bf16(false, aq[mt][1], false, bk11,
                                                   (short)0, c1, false, false);

      const v8f wv0 = mt ? wv10 : wv00;
      const v8f wv1 = mt ? wv11 : wv01;
      float sv0[8], sv1[8];
#pragma unroll
      for (int r = 0; r < 8; ++r) {
        sv0[r] = c0[r] * wv0[r];
        sv1[r] = c1[r] * wv1[r];
      }

      __bf16* p16 = (__bf16*)&ldsP32[waveInBlk][mt][0];
#pragma unroll
      for (int r = 0; r < 8; ++r) {
        float tm = fmaxf(sv0[r], sv1[r]);
        for (int msk = 8; msk >= 1; msk >>= 1)   // butterfly within half
          tm = fmaxf(tm, __shfl_xor(tm, msk, 32));
        const float newm = fmaxf(mrow[mt][r], tm);
        const float corr = __expf(mrow[mt][r] - newm);
        mrow[mt][r] = newm;
        const float p0 = __expf(sv0[r] - newm);
        const float p1 = __expf(sv1[r] - newm);
        acc[mt][0][r] *= corr; acc[mt][1][r] *= corr; acc[mt][2][r] *= corr;
        acc[mt][3][r] *= corr; acc[mt][4][r] *= corr;
        p16[(kh * 8 + r) * 32 + m]      = (__bf16)p0;   // k = n
        p16[(kh * 8 + r) * 32 + 16 + m] = (__bf16)p1;   // k = 16 + n
      }
    }

    // ---- P A-fragments from LDS (in-wave DS ordering) ---------------
    union { unsigned int u[8]; v16bf v; } ap[2];
#pragma unroll
    for (int mt = 0; mt < 2; ++mt) {
      const unsigned int* p32 = &ldsP32[waveInBlk][mt][0];
#pragma unroll
      for (int j = 0; j < 8; ++j) {
        const int kp = kh * 16 + 2 * j; // even
        ap[mt].u[j] = p32[m * 16 + (kp >> 1)];
      }
    }

    // ---- O += P x V, V fragments shared by both M-tiles -------------
#pragma unroll
    for (int ct = 0; ct < 4; ++ct) {
      const v16bf bv = vbase[(ss * 4 + ct) * 32 + lane];
      acc[0][ct] = __builtin_amdgcn_wmma_f32_16x16x32_bf16(
          false, ap[0].v, false, bv, (short)0, acc[0][ct], false, false);
      acc[1][ct] = __builtin_amdgcn_wmma_f32_16x16x32_bf16(
          false, ap[1].v, false, bv, (short)0, acc[1][ct], false, false);
    }
    acc[0][4] = __builtin_amdgcn_wmma_f32_16x16x32_bf16(
        false, ap[0].v, false, bones, (short)0, acc[0][4], false, false);
    acc[1][4] = __builtin_amdgcn_wmma_f32_16x16x32_bf16(
        false, ap[1].v, false, bones, (short)0, acc[1][4], false, false);
  }

  // ---- epilogue: divide by row sums (all columns of acc[mt][4] equal)
#pragma unroll
  for (int mt = 0; mt < 2; ++mt) {
#pragma unroll
    for (int r = 0; r < 8; ++r) {
      const int row = row0 + mt * 16 + kh * 8 + r;
      const float inv = 1.0f / acc[mt][4][r];
      out[gidx(b, row, h,      m)] = acc[mt][0][r] * inv;
      out[gidx(b, row, h, 16 + m)] = acc[mt][1][r] * inv;
      out[gidx(b, row, h, 32 + m)] = acc[mt][2][r] * inv;
      out[gidx(b, row, h, 48 + m)] = acc[mt][3][r] * inv;
    }
  }
}

// -------------------------------------------------------------------
extern "C" void kernel_launch(void* const* d_in, const int* in_sizes, int n_in,
                              void* d_out, int out_size, void* d_ws, size_t ws_size,
                              hipStream_t stream) {
  const float* Q   = (const float*)d_in[0];
  const float* K   = (const float*)d_in[1];
  const float* V   = (const float*)d_in[2];
  const float* W   = (const float*)d_in[3];
  const float* tau = (const float*)d_in[4];
  // d_in[5] = attn_mask, unused (mask_flag=False in reference)

  char* ws = (char*)d_ws;
  float* wmp = (float*)(ws + WM_OFF);
  v2bf*  qbf = (v2bf*)(ws + Q_OFF);
  v2bf*  kbf = (v2bf*)(ws + K_OFF);
  v2bf*  vbf = (v2bf*)(ws + V_OFF);

  wm_softmax_kernel<<<L_DIM, 256, 0, stream>>>(W, tau, wmp);

  // 64 bh * (64*2 or 32*4) frags * 256 pairs = 2,097,152 threads each
  packQ_kernel<<<8192, 256, 0, stream>>>(Q, qbf);
  packK_kernel<<<8192, 256, 0, stream>>>(K, kbf);
  packV_kernel<<<8192, 256, 0, stream>>>(V, vbf);

  const int waves  = B_DIM * H_DIM * (L_DIM / 32);        // 2048
  const int blocks = waves / WAVES_PER_BLOCK;             // 512
  attn_kernel<<<blocks, 32 * WAVES_PER_BLOCK, 0, stream>>>(
      (const v16bf*)qbf, (const v16bf*)kbf, (const v16bf*)vbf,
      wmp, (float*)d_out);
}
